// AttentionLePE_26688926777432
// MI455X (gfx1250) — matrix operand order repaired and checked by reading, compiled, not run on hardware
//
#include <hip/hip_runtime.h>
#include <hip/hip_bf16.h>

typedef __attribute__((ext_vector_type(16))) _Float16 v16h;
typedef __attribute__((ext_vector_type(8)))  _Float16 v8h;
typedef __attribute__((ext_vector_type(8)))  float    v8f;
typedef unsigned int u32x4 __attribute__((ext_vector_type(4)));
typedef unsigned int u32x8 __attribute__((ext_vector_type(8)));

#define B_    8
#define NH_   12
#define HD_   32
#define C_    384
#define N1_   1025
#define NP_   1056        // padded sequence length (multiple of 32)
#define CO_   1152        // 3*C
#define QSCALE 0.17677669529663689f  // 1/sqrt(32)

__device__ __forceinline__ v16h combine16(v8h lo, v8h hi) {
  v16h r;
#pragma unroll
  for (int i = 0; i < 8; ++i) { r[i] = lo[i]; r[i + 8] = hi[i]; }
  return r;
}

// max-reduce across the 16-lane group (offsets <16 never cross the half-wave)
__device__ __forceinline__ float redmax16(float t) {
#pragma unroll
  for (int off = 1; off < 16; off <<= 1) t = fmaxf(t, __shfl_xor(t, off));
  return t;
}

__device__ __forceinline__ v8f wmma_f16(v16h a, v16h b, v8f c) {
  return __builtin_amdgcn_wmma_f32_16x16x32_f16(false, a, false, b,
                                                (short)0, c, false, false);
}

// ---------------------------------------------------------------- TDM helpers
// D# group1 for a 2D (tile_dim2=0) f16 tensor: dims/strides in elements.
__device__ __forceinline__ u32x8 tdm_g1_2d(unsigned t_dim0, unsigned t_dim1,
                                           unsigned tile0, unsigned tile1,
                                           unsigned stride0) {
  u32x8 g;
  g[0] = 1u << 16;                                       // data_size = 2 bytes
  g[1] = (t_dim0 & 0xFFFFu) << 16;                       // tensor_dim0[15:0]
  g[2] = (t_dim0 >> 16) | ((t_dim1 & 0xFFFFu) << 16);    // dim0[31:16] | dim1[15:0]
  g[3] = (t_dim1 >> 16) | (tile0 << 16);                 // dim1[31:16] | tile_dim0
  g[4] = tile1;                                          // tile_dim1 (tile_dim2=0)
  g[5] = stride0;                                        // tensor_dim0_stride[31:0]
  g[6] = 0u;
  g[7] = 0u;
  return g;
}

// Issue TENSOR_LOAD_TO_LDS (2-group form, 2D tensor). lds_addr = wave-relative
// LDS byte address (low 32 bits of the generic shared pointer).
__device__ __forceinline__ void tdm_load_2d(unsigned lds_addr, const void* gaddr,
                                            u32x8 g1) {
  unsigned long long ga = (unsigned long long)gaddr;
  u32x4 g0;
  g0[0] = 1u;                                            // count=1, user mode
  g0[1] = lds_addr;                                      // lds_addr
  g0[2] = (unsigned)ga;                                  // global_addr[31:0]
  g0[3] = ((unsigned)(ga >> 32) & 0x01FFFFFFu) | (2u << 30); // addr[56:32]|type=2
  asm volatile("tensor_load_to_lds %0, %1" :: "s"(g0), "s"(g1) : "memory");
}

__device__ __forceinline__ unsigned lds_off(const void* p) {
  return (unsigned)(unsigned long long)p;   // LDS aperture: offset in [31:0]
}

// ---------------------------------------------------------------- conversions
__global__ void k_build_xcat(const float* __restrict__ x,
                             const float* __restrict__ cls,
                             _Float16* __restrict__ xc) {
  int i = blockIdx.x * 256 + threadIdx.x;           // over B*NP*C
  if (i >= B_ * NP_ * C_) return;
  int c = i % C_;
  int t = (i / C_) % NP_;
  int b = i / (C_ * NP_);
  float v = 0.f;
  if (t == 0)          v = cls[b * C_ + c];
  else if (t <= 1024)  v = x[((size_t)(b * 1024 + (t - 1))) * C_ + c];
  xc[i] = (_Float16)v;
}

__global__ void k_f32_to_f16(const float* __restrict__ s,
                             _Float16* __restrict__ d, int n) {
  int i = blockIdx.x * 256 + threadIdx.x;
  if (i < n) d[i] = (_Float16)s[i];
}

// ------------------------------------------------------------------ QKV GEMM
// block = 4 waves; block tile 64 rows x 64 cols; wave tile 16x64 (acc[4] only,
// avoids accumulator spills). B (weight) k-slices staged in LDS by TDM.
__global__ void k_qkv_gemm(const _Float16* __restrict__ xc,
                           const _Float16* __restrict__ w,
                           _Float16* __restrict__ Qh,
                           _Float16* __restrict__ Kh,
                           _Float16* __restrict__ Vt) {
  __shared__ __align__(64) _Float16 sB[2][64 * 32];   // [n_local][k(32)]
  const int tid  = threadIdx.x;
  const int wave = tid >> 5;
  const int lane = tid & 31;
  const int l15  = lane & 15;
  const int hs   = lane >> 4;
  const int R0   = blockIdx.x * 64 + wave * 16;       // wave's M-tile
  const int n0   = blockIdx.y * 64;

  const u32x8 g1 = tdm_g1_2d(C_, CO_, 32, 64, C_);
  if (wave == 0) {
    tdm_load_2d(lds_off(&sB[0][0]), w + (size_t)n0 * C_, g1);
    __builtin_amdgcn_s_wait_tensorcnt(0);
  }
  __syncthreads();

  v8f acc[4];
#pragma unroll
  for (int j = 0; j < 4; ++j) acc[j] = 0.f;

  const _Float16* arow = xc + (size_t)(R0 + l15) * C_;

  for (int ks = 0; ks < C_ / 32; ++ks) {
    const int k0  = ks * 32;
    const int cur = ks & 1;
    if (wave == 0 && ks + 1 < C_ / 32)
      tdm_load_2d(lds_off(&sB[cur ^ 1][0]), w + (size_t)n0 * C_ + (k0 + 32), g1);

    v16h a = combine16(*(const v8h*)(arow + k0 + hs * 16),
                       *(const v8h*)(arow + k0 + hs * 16 + 8));
#pragma unroll
    for (int j = 0; j < 4; ++j) {
      v16h b = *(const v16h*)(&sB[cur][(16 * j + l15) * 32 + hs * 16]);
      acc[j] = wmma_f16(a, b, acc[j]);
    }
    if (wave == 0 && ks + 1 < C_ / 32) __builtin_amdgcn_s_wait_tensorcnt(0);
    __syncthreads();
  }

#pragma unroll
  for (int j = 0; j < 4; ++j) {
    int n = n0 + 16 * j + l15;
    int p = n / C_;                 // 0=q 1=k 2=v
    int c = n % C_;
    int h = c >> 5;
    int d = c & 31;
#pragma unroll
    for (int r = 0; r < 8; ++r) {
      int m = R0 + r + hs * 8;
      int b = m / NP_;
      int t = m % NP_;
      float v = acc[j][r];
      size_t tok = (((size_t)b * NH_ + h) * NP_ + t) * HD_ + d;
      if (p == 0)      Qh[tok] = (_Float16)(v * QSCALE);
      else if (p == 1) Kh[tok] = (_Float16)v;
      else             Vt[(((size_t)b * NH_ + h) * HD_ + d) * NP_ + t] = (_Float16)v;
    }
  }
}

// ------------------------------------------------------------ flash attention
// block = 4 waves = 4 consecutive q-tiles of one (b,h); K/V chunks staged in
// LDS by TDM (double-buffered); row-sum via WMMA against all-ones B.
__global__ void k_attn(const _Float16* __restrict__ Qh,
                       const _Float16* __restrict__ Kh,
                       const _Float16* __restrict__ Vt,
                       float* __restrict__ attf) {
  __shared__ __align__(64) _Float16 sK[2][32 * 32];   // [t_local][d]
  __shared__ __align__(64) _Float16 sV[2][32 * 32];   // [d][t_local]
  __shared__ __align__(64) _Float16 sP[4][16 * 32];   // per-wave P bounce
  const int tid  = threadIdx.x;
  const int wave = tid >> 5;
  const int lane = tid & 31;
  const int l15  = lane & 15;
  const int hs   = lane >> 4;

  int id = blockIdx.x;                 // B*NH*17 blocks
  int g  = id % 17;
  int h  = (id / 17) % NH_;
  int b  = id / (17 * NH_);

  const _Float16* Qb = Qh + ((size_t)b * NH_ + h) * NP_ * HD_;
  const _Float16* Kb = Kh + ((size_t)b * NH_ + h) * NP_ * HD_;
  const _Float16* Vb = Vt + ((size_t)b * NH_ + h) * HD_ * NP_;

  int qt = g * 4 + wave;
  if (qt > 65) qt = 65;                // clamped waves compute zeros, store nothing
  int qrow = qt * 16 + l15;
  v16h aq = combine16(*(const v8h*)(Qb + (size_t)qrow * HD_ + hs * 16),
                      *(const v8h*)(Qb + (size_t)qrow * HD_ + hs * 16 + 8));

  v16h ones;
#pragma unroll
  for (int i = 0; i < 16; ++i) ones[i] = (_Float16)1.0f;

  const u32x8 g1K = tdm_g1_2d(HD_, NP_, HD_, 32, HD_);   // tile 32d x 32t
  const u32x8 g1V = tdm_g1_2d(NP_, HD_, 32, HD_, NP_);   // tile 32t x 32d

  if (wave == 0) {
    tdm_load_2d(lds_off(&sK[0][0]), Kb, g1K);
    tdm_load_2d(lds_off(&sV[0][0]), Vb, g1V);
    __builtin_amdgcn_s_wait_tensorcnt(0);
  }
  __syncthreads();

  v8f o0 = 0.f, o1 = 0.f, mrow, lrow = 0.f;
#pragma unroll
  for (int r = 0; r < 8; ++r) mrow[r] = -1e30f;

  for (int j = 0; j < NP_ / 32; ++j) {
    const int kv0 = j * 32;
    const int cur = j & 1;
    if (wave == 0 && j + 1 < NP_ / 32) {
      tdm_load_2d(lds_off(&sK[cur ^ 1][0]), Kb + (size_t)(kv0 + 32) * HD_, g1K);
      tdm_load_2d(lds_off(&sV[cur ^ 1][0]), Vb + (kv0 + 32), g1V);
    }

    v16h bk0 = *(const v16h*)(&sK[cur][(l15) * HD_ + hs * 16]);
    v16h bk1 = *(const v16h*)(&sK[cur][(16 + l15) * HD_ + hs * 16]);
    v8f z = 0.f;
    v8f s0 = wmma_f16(aq, bk0, z);
    v8f s1 = wmma_f16(aq, bk1, z);

    if (kv0 + 31 >= N1_) {             // mask padded KV columns (uniform branch)
      bool m0 = (kv0 + l15)      >= N1_;
      bool m1 = (kv0 + 16 + l15) >= N1_;
#pragma unroll
      for (int r = 0; r < 8; ++r) {
        if (m0) s0[r] = -1e30f;
        if (m1) s1[r] = -1e30f;
      }
    }

    v8f mnew, alpha, p0, p1;
#pragma unroll
    for (int r = 0; r < 8; ++r) {
      float mc = redmax16(fmaxf(s0[r], s1[r]));
      mnew[r]  = fmaxf(mrow[r], mc);
      alpha[r] = __expf(mrow[r] - mnew[r]);
      mrow[r]  = mnew[r];
      p0[r] = __expf(s0[r] - mnew[r]);
      p1[r] = __expf(s1[r] - mnew[r]);
      o0[r] *= alpha[r];
      o1[r] *= alpha[r];
    }

    // C-layout -> A-layout transpose of P through per-wave LDS
#pragma unroll
    for (int r = 0; r < 8; ++r) {
      int row = r + hs * 8;
      sP[wave][row * 32 + l15]      = (_Float16)p0[r];
      sP[wave][row * 32 + 16 + l15] = (_Float16)p1[r];
    }
    asm volatile("s_wait_dscnt 0" ::: "memory");
    v16h ap = combine16(*(const v8h*)(&sP[wave][l15 * 32 + hs * 16]),
                        *(const v8h*)(&sP[wave][l15 * 32 + hs * 16 + 8]));

    // row-sum of P via WMMA against all-ones B (broadcast in C-layout)
    v8f lsum = wmma_f16(ap, ones, z);
#pragma unroll
    for (int r = 0; r < 8; ++r) lrow[r] = lrow[r] * alpha[r] + lsum[r];

    v16h bv0 = *(const v16h*)(&sV[cur][(l15) * 32 + hs * 16]);
    v16h bv1 = *(const v16h*)(&sV[cur][(16 + l15) * 32 + hs * 16]);
    o0 = wmma_f16(ap, bv0, o0);
    o1 = wmma_f16(ap, bv1, o1);

    if (wave == 0 && j + 1 < NP_ / 32) __builtin_amdgcn_s_wait_tensorcnt(0);
    __syncthreads();
  }

#pragma unroll
  for (int r = 0; r < 8; ++r) {
    int t = qt * 16 + r + hs * 8;
    if (t < N1_) {
      float inv = 1.0f / lrow[r];
      size_t base = ((size_t)b * NP_ + t) * C_ + h * 32;
      attf[base + l15]      = o0[r] * inv;
      attf[base + 16 + l15] = o1[r] * inv;
    }
  }
}

// ----------------------------------------------- LePE depthwise conv + fusion
__global__ void k_lepe_fuse(const float* __restrict__ x,
                            const float* __restrict__ attf,
                            const float* __restrict__ lw,
                            const float* __restrict__ lb,
                            _Float16* __restrict__ zh) {
  int i = blockIdx.x * 256 + threadIdx.x;           // over B*NP*C
  if (i >= B_ * NP_ * C_) return;
  int c = i % C_;
  int t = (i / C_) % NP_;
  int b = i / (C_ * NP_);
  float v = 0.f;
  if (t == 0) {
    v = attf[(size_t)b * NP_ * C_ + c];
  } else if (t <= 1024) {
    int s = t - 1, y = s >> 5, xq = s & 31;
    float acc = lb[c];
#pragma unroll
    for (int ky = 0; ky < 5; ++ky) {
      int yy = y + ky - 2;
      if (yy < 0 || yy >= 32) continue;
#pragma unroll
      for (int kx = 0; kx < 5; ++kx) {
        int xx = xq + kx - 2;
        if (xx < 0 || xx >= 32) continue;
        acc += x[((size_t)(b * 32 + yy) * 32 + xx) * C_ + c] * lw[c * 25 + ky * 5 + kx];
      }
    }
    v = attf[((size_t)b * NP_ + t) * C_ + c] + acc;
  }
  zh[i] = (_Float16)v;
}

// ------------------------------------------------------------------ Proj GEMM
__global__ void k_proj_gemm(const _Float16* __restrict__ zh,
                            const _Float16* __restrict__ w,
                            const float* __restrict__ bias,
                            float* __restrict__ out) {
  __shared__ __align__(64) _Float16 sB[2][64 * 32];
  const int tid  = threadIdx.x;
  const int wave = tid >> 5;
  const int lane = tid & 31;
  const int l15  = lane & 15;
  const int hs   = lane >> 4;
  const int R0   = blockIdx.x * 64 + wave * 16;
  const int n0   = blockIdx.y * 64;

  const u32x8 g1 = tdm_g1_2d(C_, C_, 32, 64, C_);
  if (wave == 0) {
    tdm_load_2d(lds_off(&sB[0][0]), w + (size_t)n0 * C_, g1);
    __builtin_amdgcn_s_wait_tensorcnt(0);
  }
  __syncthreads();

  v8f acc[4];
#pragma unroll
  for (int j = 0; j < 4; ++j) acc[j] = 0.f;

  const _Float16* arow = zh + (size_t)(R0 + l15) * C_;

  for (int ks = 0; ks < C_ / 32; ++ks) {
    const int k0  = ks * 32;
    const int cur = ks & 1;
    if (wave == 0 && ks + 1 < C_ / 32)
      tdm_load_2d(lds_off(&sB[cur ^ 1][0]), w + (size_t)n0 * C_ + (k0 + 32), g1);

    v16h a = combine16(*(const v8h*)(arow + k0 + hs * 16),
                       *(const v8h*)(arow + k0 + hs * 16 + 8));
#pragma unroll
    for (int j = 0; j < 4; ++j) {
      v16h b = *(const v16h*)(&sB[cur][(16 * j + l15) * 32 + hs * 16]);
      acc[j] = wmma_f16(a, b, acc[j]);
    }
    if (wave == 0 && ks + 1 < C_ / 32) __builtin_amdgcn_s_wait_tensorcnt(0);
    __syncthreads();
  }

  const size_t cls_base = (size_t)B_ * 1024 * C_;
#pragma unroll
  for (int j = 0; j < 4; ++j) {
    int n = n0 + 16 * j + l15;
    float bn = bias[n];
#pragma unroll
    for (int r = 0; r < 8; ++r) {
      int m = R0 + r + hs * 8;
      int b = m / NP_;
      int t = m % NP_;
      float v = acc[j][r] + bn;
      if (t == 0)          out[cls_base + (size_t)b * C_ + n] = v;
      else if (t <= 1024)  out[((size_t)(b * 1024 + (t - 1))) * C_ + n] = v;
    }
  }
}

// ------------------------------------------------------------------- launcher
extern "C" void kernel_launch(void* const* d_in, const int* in_sizes, int n_in,
                              void* d_out, int out_size, void* d_ws, size_t ws_size,
                              hipStream_t stream) {
  (void)in_sizes; (void)n_in; (void)out_size; (void)ws_size;
  const float* x      = (const float*)d_in[0];
  const float* cls    = (const float*)d_in[1];
  const float* qkv_w  = (const float*)d_in[2];
  const float* proj_w = (const float*)d_in[3];
  const float* proj_b = (const float*)d_in[4];
  const float* lepe_w = (const float*)d_in[5];
  const float* lepe_b = (const float*)d_in[6];
  float* out = (float*)d_out;

  char* ws = (char*)d_ws;
  const size_t SZ_XCAT = (size_t)B_ * NP_ * C_ * 2;        // 6488064
  const size_t SZ_QKVW = (size_t)CO_ * C_ * 2;             //  884736
  const size_t SZ_PRJW = (size_t)C_ * C_ * 2;              //  294912
  const size_t SZ_HEAD = (size_t)B_ * NH_ * NP_ * HD_ * 2; // 6488064
  const size_t SZ_ATTF = (size_t)B_ * NP_ * C_ * 4;        // 12976128

  size_t off = 0;
  _Float16* xcat_h = (_Float16*)(ws + off); off += SZ_XCAT;
  _Float16* qkvw_h = (_Float16*)(ws + off); off += SZ_QKVW;
  _Float16* projw_h= (_Float16*)(ws + off); off += SZ_PRJW;
  _Float16* Qh     = (_Float16*)(ws + off); off += SZ_HEAD;
  _Float16* Kh     = (_Float16*)(ws + off); off += SZ_HEAD;
  _Float16* Vth    = (_Float16*)(ws + off); off += SZ_HEAD;
  float*    attf   = (float*)   (ws + off); off += SZ_ATTF;
  _Float16* zh     = (_Float16*)(ws + off); off += SZ_XCAT;

  const int TOT = B_ * NP_ * C_;                           // 3244032
  k_build_xcat<<<(TOT + 255) / 256, 256, 0, stream>>>(x, cls, xcat_h);
  k_f32_to_f16<<<(CO_ * C_ + 255) / 256, 256, 0, stream>>>(qkv_w, qkvw_h, CO_ * C_);
  k_f32_to_f16<<<(C_ * C_ + 255) / 256, 256, 0, stream>>>(proj_w, projw_h, C_ * C_);

  k_qkv_gemm<<<dim3(B_ * NP_ / 64, CO_ / 64), 128, 0, stream>>>(xcat_h, qkvw_h, Qh, Kh, Vth);
  k_attn<<<B_ * NH_ * 17, 128, 0, stream>>>(Qh, Kh, Vth, attf);
  k_lepe_fuse<<<(TOT + 255) / 256, 256, 0, stream>>>(x, attf, lepe_w, lepe_b, zh);
  k_proj_gemm<<<dim3(B_ * NP_ / 64, C_ / 64), 128, 0, stream>>>(zh, projw_h, proj_b, out);
}
